// NodeClassifier_22325240005082
// MI455X (gfx1250) — compile-verified, matrix operand order checked by reading, not run on hardware
//
#include <hip/hip_runtime.h>
#include <hip/hip_bf16.h>

typedef __attribute__((ext_vector_type(16))) __bf16 v16bf;
typedef __attribute__((ext_vector_type(8)))  __bf16 v8bf;
typedef __attribute__((ext_vector_type(8)))  float  v8f;

#define NNODES 100000
#define NEDGES 3200000

// One-shot: W[K,N] f32 -> Wt[N,K] bf16 (transposed, K contiguous)
__global__ __launch_bounds__(256) void convert_wt(
    const float* __restrict__ W, __bf16* __restrict__ Wt, int K, int N)
{
    int idx = blockIdx.x * 256 + threadIdx.x;
    if (idx >= K * N) return;
    int k = idx / N;
    int n = idx - k * N;
    Wt[(size_t)n * K + k] = (__bf16)W[idx];
}

// A-fragment loaders: 16 bf16 per lane, ISA 16-bit A layout
// VGPR0-3: k = half*8 + 2v ; VGPR4-7: k = 16 + half*8 + 2(v-4)
// p points at row start + half*8 + ks.
__device__ __forceinline__ v16bf load_afrag(const float* __restrict__ p)
{
    const float4 a0 = *reinterpret_cast<const float4*>(p);
    const float4 a1 = *reinterpret_cast<const float4*>(p + 4);
    const float4 a2 = *reinterpret_cast<const float4*>(p + 16);
    const float4 a3 = *reinterpret_cast<const float4*>(p + 20);
    v16bf af;
    af[0]  = (__bf16)a0.x; af[1]  = (__bf16)a0.y; af[2]  = (__bf16)a0.z; af[3]  = (__bf16)a0.w;
    af[4]  = (__bf16)a1.x; af[5]  = (__bf16)a1.y; af[6]  = (__bf16)a1.z; af[7]  = (__bf16)a1.w;
    af[8]  = (__bf16)a2.x; af[9]  = (__bf16)a2.y; af[10] = (__bf16)a2.z; af[11] = (__bf16)a2.w;
    af[12] = (__bf16)a3.x; af[13] = (__bf16)a3.y; af[14] = (__bf16)a3.z; af[15] = (__bf16)a3.w;
    return af;
}

__device__ __forceinline__ v16bf load_afrag(const __bf16* __restrict__ p)
{
    const v8bf a0 = *reinterpret_cast<const v8bf*>(p);        // k in [half*8, +8)
    const v8bf a1 = *reinterpret_cast<const v8bf*>(p + 16);   // k in [16+half*8, +8)
    v16bf af;
    #pragma unroll
    for (int j = 0; j < 8; ++j) { af[j] = a0[j]; af[8 + j] = a1[j]; }
    return af;
}

// ---------------------------------------------------------------------------
// WMMA GEMM:  C[M,Nn] = act( A[M,K] @ W (+ bias) ),  W pre-transposed bf16
// Wt[Nn][K]. Block: 256(M) x 64(N), 8 waves; wave: 32M x 64N (two A frags,
// 8 accumulators: each B fragment feeds two v_wmma). W tile double-buffered
// in LDS [n][k] stride 56 (112B rows, b128-aligned): one b128 reg-prefetch +
// one ds_store + ONE barrier per K-step. A: f32 (cvt on load) or bf16.
// C: f32 or bf16. Out-of-range rows/cols index-clamped, masked at store.
// ---------------------------------------------------------------------------
template <typename AT, typename CT>
__global__ __launch_bounds__(256) void gemm_bf16_wmma(
    const AT* __restrict__ A, const __bf16* __restrict__ Wt,
    const float* __restrict__ bias, CT* __restrict__ C,
    int M, int K, int Nn, int do_relu)
{
    __shared__ __bf16 Wlds[2][64 * 56];

    const int lane = threadIdx.x & 31;
    const int wave = threadIdx.x >> 5;
    const int half = lane >> 4;     // ISA: lanes 0-15 / 16-31 K-half split
    const int l16  = lane & 15;

    const int m0 = blockIdx.x * 256 + wave * 32;
    const int n0 = blockIdx.y * 64;

    int r0 = m0 + l16;       if (r0 > M - 1) r0 = M - 1;   // clamp; masked at store
    int r1 = m0 + 16 + l16;  if (r1 > M - 1) r1 = M - 1;
    const AT* A0 = A + (size_t)r0 * K + half * 8;
    const AT* A1 = A + (size_t)r1 * K + half * 8;

    // cooperative W-stage: thread owns column (n0+wn), k-chunk [wkq*8, wkq*8+8)
    const int wn  = (int)threadIdx.x & 63;
    const int wkq = (int)threadIdx.x >> 6;                  // 0..3
    int gn = n0 + wn; if (gn > Nn - 1) gn = Nn - 1;         // clamp, no predicate
    const __bf16* wsrc = Wt + (size_t)gn * K + wkq * 8;     // loop-invariant base
    const int wofs = wn * 56 + wkq * 8;

    v8f acc0[4] = {};
    v8f acc1[4] = {};

    const int nt = K >> 5;                                  // K-steps of 32
    v8bf wreg = *reinterpret_cast<const v8bf*>(wsrc);       // tile 0 in regs

    for (int t = 0; t < nt; ++t) {
        const int ks  = t << 5;
        const int cur = t & 1;

        *reinterpret_cast<v8bf*>(&Wlds[cur][wofs]) = wreg;  // stage tile t
        if (t + 1 < nt)                                     // reg-prefetch tile t+1
            wreg = *reinterpret_cast<const v8bf*>(wsrc + (ks + 32));
        __syncthreads();                                    // single barrier / step

        if (t + 1 < nt) {
            __builtin_prefetch(A0 + ks + 32, 0, 1);         // global_prefetch_b8
            __builtin_prefetch(A1 + ks + 32, 0, 1);
        }

        const v16bf af0 = load_afrag(A0 + ks);
        const v16bf af1 = load_afrag(A1 + ks);

        #pragma unroll
        for (int f = 0; f < 4; ++f) {
            // B fragment: lane holds column (f*16+l16), same K-pair pattern as A
            const __bf16* bp = &Wlds[cur][(f * 16 + l16) * 56 + half * 8];
            v8bf b0 = *reinterpret_cast<const v8bf*>(bp);
            v8bf b1 = *reinterpret_cast<const v8bf*>(bp + 16);
            v16bf bfr;
            #pragma unroll
            for (int j = 0; j < 8; ++j) { bfr[j] = b0[j]; bfr[8 + j] = b1[j]; }
            acc0[f] = __builtin_amdgcn_wmma_f32_16x16x32_bf16(
                false, af0, false, bfr, (short)0, acc0[f], false, false);
            acc1[f] = __builtin_amdgcn_wmma_f32_16x16x32_bf16(
                false, af1, false, bfr, (short)0, acc1[f], false, false);
        }
    }

    // Epilogue: C layout — lane: n = l16, VGPR v: m = half*8 + v
    #pragma unroll
    for (int f = 0; f < 4; ++f) {
        int n = n0 + f * 16 + l16;
        if (n >= Nn) continue;
        float bv = bias ? bias[n] : 0.f;
        #pragma unroll
        for (int v = 0; v < 8; ++v) {
            int m = m0 + half * 8 + v;
            if (m < M) {
                float val = acc0[f][v] + bv;
                if (do_relu) val = fmaxf(val, 0.f);
                C[(size_t)m * Nn + n] = (CT)val;
            }
            int m2 = m + 16;
            if (m2 < M) {
                float val = acc1[f][v] + bv;
                if (do_relu) val = fmaxf(val, 0.f);
                C[(size_t)m2 * Nn + n] = (CT)val;
            }
        }
    }
}

// ---------------------------------------------------------------------------
// Edge-parallel SpMM scatter (F = 256): 64 threads per edge, float4 chunks.
// out[dst] += w[e] * h[src]   via global_atomic_add_f32 (L2-resident RMW).
// ---------------------------------------------------------------------------
__global__ __launch_bounds__(256) void spmm_edges_f256(
    const int* __restrict__ esrc, const int* __restrict__ edst,
    const float* __restrict__ ew, const float* __restrict__ h,
    float* __restrict__ out)
{
    long long idx = (long long)blockIdx.x * 256 + threadIdx.x;
    if (idx >= (long long)NEDGES * 64) return;
    int e = (int)(idx >> 6);
    int c = (int)(idx & 63) << 2;
    int s = esrc[e];
    int d = edst[e];
    float w = ew[e];
    const float4 v = *reinterpret_cast<const float4*>(h + (size_t)s * 256 + c);
    float* o = out + (size_t)d * 256 + c;
    atomicAdd(o + 0, w * v.x);
    atomicAdd(o + 1, w * v.y);
    atomicAdd(o + 2, w * v.z);
    atomicAdd(o + 3, w * v.w);
}

__global__ __launch_bounds__(256) void fill_zero(float* __restrict__ p, long long n)
{
    long long i = (long long)blockIdx.x * 256 + threadIdx.x;
    if (i < n) p[i] = 0.f;
}

// post-spmm bias (+optional relu), F == 256
__global__ __launch_bounds__(256) void bias_act256(
    float* __restrict__ h, const float* __restrict__ b, long long total, int do_relu)
{
    long long i = (long long)blockIdx.x * 256 + threadIdx.x;
    if (i >= total) return;
    float v = h[i] + b[(int)(i & 255)];
    if (do_relu) v = fmaxf(v, 0.f);
    h[i] = v;
}

// in-place row softmax over 40 classes
__global__ __launch_bounds__(256) void softmax40(float* __restrict__ p, int M)
{
    int i = blockIdx.x * 256 + threadIdx.x;
    if (i >= M) return;
    float* row = p + (size_t)i * 40;
    float mx = row[0];
    #pragma unroll
    for (int c = 1; c < 40; ++c) mx = fmaxf(mx, row[c]);
    float e[40];
    float s = 0.f;
    #pragma unroll
    for (int c = 0; c < 40; ++c) { e[c] = __expf(row[c] - mx); s += e[c]; }
    float inv = 1.f / s;
    #pragma unroll
    for (int c = 0; c < 40; ++c) row[c] = e[c] * inv;
}

extern "C" void kernel_launch(void* const* d_in, const int* in_sizes, int n_in,
                              void* d_out, int out_size, void* d_ws, size_t ws_size,
                              hipStream_t stream) {
    (void)in_sizes; (void)n_in; (void)out_size; (void)ws_size;
    const float* x    = (const float*)d_in[0];
    const int*   esrc = (const int*)  d_in[1];
    const int*   edst = (const int*)  d_in[2];
    const float* ew   = (const float*)d_in[3];
    const float* w1   = (const float*)d_in[4];
    const float* b1   = (const float*)d_in[5];
    const float* w2   = (const float*)d_in[6];
    const float* b2   = (const float*)d_in[7];
    const float* wc1  = (const float*)d_in[8];
    const float* bc1  = (const float*)d_in[9];
    const float* wc2  = (const float*)d_in[10];
    const float* bc2  = (const float*)d_in[11];
    const float* wc3  = (const float*)d_in[12];
    const float* bc3  = (const float*)d_in[13];
    const float* wc4  = (const float*)d_in[14];
    const float* bc4  = (const float*)d_in[15];
    float* out = (float*)d_out;

    float* bufA = (float*)d_ws;                       // [N,512] f32  (204.8 MB)
    float* bufB = bufA + (size_t)NNODES * 512;        // [N,512] f32  (204.8 MB)
    __bf16* hA16 = (__bf16*)bufA;                     // bf16 aliases for MLP chain
    __bf16* hB16 = (__bf16*)bufB;
    // pre-transposed bf16 weights (1.7 MB total) after the ping-pong buffers
    __bf16* wt1  = (__bf16*)(bufB + (size_t)NNODES * 512);
    __bf16* wt2  = wt1  + 512 * 256;
    __bf16* wtc1 = wt2  + 256 * 256;
    __bf16* wtc2 = wtc1 + 256 * 512;
    __bf16* wtc3 = wtc2 + 512 * 512;
    __bf16* wtc4 = wtc3 + 512 * 512;

    const dim3 blk(256);
    const int gmx = (NNODES + 255) / 256;             // 391
    const long long nel256 = (long long)NNODES * 256;
    const unsigned fillBlocks  = (unsigned)((nel256 + 255) / 256);
    const unsigned spmmBlocks  = (unsigned)(((long long)NEDGES * 64 + 255) / 256);
    const unsigned baBlocks    = fillBlocks;
    const unsigned smBlocks    = (NNODES + 255) / 256;

    // ---- one-shot weight convert+transpose to bf16 [N][K] ----
    convert_wt<<<(512*256 + 255)/256, blk, 0, stream>>>(w1,  wt1,  512, 256);
    convert_wt<<<(256*256 + 255)/256, blk, 0, stream>>>(w2,  wt2,  256, 256);
    convert_wt<<<(256*512 + 255)/256, blk, 0, stream>>>(wc1, wtc1, 256, 512);
    convert_wt<<<(512*512 + 255)/256, blk, 0, stream>>>(wc2, wtc2, 512, 512);
    convert_wt<<<(512*512 + 255)/256, blk, 0, stream>>>(wc3, wtc3, 512, 512);
    convert_wt<<<(512*40  + 255)/256, blk, 0, stream>>>(wc4, wtc4, 512, 40);

    // ---- GCN layer 1: h1 = relu(spmm(x@w1) + b1) ----  (f32 path, spmm needs f32)
    gemm_bf16_wmma<float,float><<<dim3(gmx, 4), blk, 0, stream>>>(x, wt1, nullptr, bufA, NNODES, 512, 256, 0);
    fill_zero<<<fillBlocks, blk, 0, stream>>>(bufB, nel256);
    spmm_edges_f256<<<spmmBlocks, blk, 0, stream>>>(esrc, edst, ew, bufA, bufB);
    bias_act256<<<baBlocks, blk, 0, stream>>>(bufB, b1, nel256, 1);

    // ---- GCN layer 2: h2 = spmm(h1@w2) + b2 ----
    gemm_bf16_wmma<float,float><<<dim3(gmx, 4), blk, 0, stream>>>(bufB, wt2, nullptr, bufA, NNODES, 256, 256, 0);
    fill_zero<<<fillBlocks, blk, 0, stream>>>(bufB, nel256);
    spmm_edges_f256<<<spmmBlocks, blk, 0, stream>>>(esrc, edst, ew, bufA, bufB);
    bias_act256<<<baBlocks, blk, 0, stream>>>(bufB, b2, nel256, 0);

    // ---- classifier MLP (bf16 activations between GEMMs) ----
    gemm_bf16_wmma<float,__bf16><<<dim3(gmx, 8), blk, 0, stream>>>(bufB, wtc1, bc1, hA16, NNODES, 256, 512, 1);
    gemm_bf16_wmma<__bf16,__bf16><<<dim3(gmx, 8), blk, 0, stream>>>(hA16, wtc2, bc2, hB16, NNODES, 512, 512, 1);
    gemm_bf16_wmma<__bf16,__bf16><<<dim3(gmx, 8), blk, 0, stream>>>(hB16, wtc3, bc3, hA16, NNODES, 512, 512, 1);
    gemm_bf16_wmma<__bf16,float><<<dim3(gmx, 1), blk, 0, stream>>>(hA16, wtc4, bc4, out, NNODES, 512, 40, 0);

    // ---- softmax over 40 classes, in place on d_out ----
    softmax40<<<smBlocks, blk, 0, stream>>>(out, NNODES);
}
